// ChebConv_66881230733430
// MI455X (gfx1250) — matrix operand order repaired and checked by reading, not compiled, not run on hardware
//
#include <hip/hip_runtime.h>
#include <hip/hip_bf16.h>

typedef __attribute__((ext_vector_type(4)))  float  v4f;
typedef __attribute__((ext_vector_type(8)))  float  v8f;
typedef __attribute__((ext_vector_type(8)))  __bf16 v8bf;
typedef __attribute__((ext_vector_type(16))) __bf16 v16bf;

#define BATCH   8192
#define NPT     9
#define INC     1024
#define OUTC    1024

// GEMM tiling: M_tile=144 rows (16 batches * 9), N_tile=128, K_tile=64, K_total=3072
#define BG      16
#define MT      144
#define NT      128
#define KT      64

// LDS layout (bytes). All offsets 16B aligned.
#define CH_STRIDE  12                          // cheb row padded 9->12 floats
#define CHEB_BYTES (3*MT*CH_STRIDE*4)          // 20736
#define XS_STRIDE  12                          // Xs transposed [b][c][j], padded floats
#define XS_BYTES   (BG*KT*XS_STRIDE*4)         // 49152
#define AB_STRIDE  72                          // bf16 row stride (64 -> 72): conflict-free b128 frags
#define AHL_BYTES  (MT*AB_STRIDE*2)            // 20736 each (hi, lo)
#define WHL_BYTES  (NT*AB_STRIDE*2)            // 18432 each (hi, lo)
#define SMEM_BYTES (CHEB_BYTES + XS_BYTES + 2*AHL_BYTES + 2*WHL_BYTES)  // 148224

// f32 -> bf16 (RNE) bit helpers: avoid reliance on scalar __bf16 conversions.
static __device__ __forceinline__ unsigned short bf16h(float f) {
    unsigned u = __float_as_uint(f);
    u += 0x7FFFu + ((u >> 16) & 1u);
    return (unsigned short)(u >> 16);
}
static __device__ __forceinline__ float bf16f(unsigned short h) {
    return __uint_as_float(((unsigned)h) << 16);
}

// ---------------------------------------------------------------------------
// Kernel 1: cheb polynomials in RAW [B][3][9][9] layout (== flat [3B][81]).
// T0 = I, T1 = L, T2 = 2*L@L - I.
// ---------------------------------------------------------------------------
__global__ __launch_bounds__(256) void cheb_kernel(const float* __restrict__ mulL,
                                                   float* __restrict__ chebraw) {
    __shared__ float Ls[81];
    const int b = blockIdx.x;
    const int t = threadIdx.x;
    if (t < 81) Ls[t] = mulL[b * 81 + t];
    __syncthreads();
    if (t < 243) {
        const int kk = t / 81;
        const int r  = t % 81;
        const int i  = r / 9;
        const int j  = r % 9;
        float v;
        if (kk == 0) {
            v = (i == j) ? 1.0f : 0.0f;
        } else if (kk == 1) {
            v = Ls[r];
        } else {
            float s = 0.0f;
#pragma unroll
            for (int m = 0; m < 9; ++m) s += Ls[i * 9 + m] * Ls[m * 9 + j];
            v = 2.0f * s - ((i == j) ? 1.0f : 0.0f);
        }
        chebraw[b * 243 + t] = v;
    }
}

// ---------------------------------------------------------------------------
// Kernel 2: fused A-build + BF16x3 split-precision GEMM.
//   out[row, n] = sum_k A_all[row, k] * Wstack[k, n] + bias[n]
// A_all[row, kp*1024+c] = sum_j cheb(kp,row)[j] * X[b'][j, c]
// Split: v = hi + lo (bf16 pair); acc += Ah*Bh + Ah*Bl + Al*Bh (fp32 acc).
// Block: 288 threads = 9 waves; wave w owns C rows [16w,16w+16) x 128 cols.
// ---------------------------------------------------------------------------
__global__ __launch_bounds__(288, 1) void cheb_gemm_kernel(
        const float* __restrict__ X,       // [8192][9][1024]
        const float* __restrict__ W,       // [3][1][1024][1024]
        const float* __restrict__ bias,    // [1024]
        const float* __restrict__ chebraw, // [3*8192][81]
        float* __restrict__ out) {         // [8192*9][1024]
    extern __shared__ __align__(16) char smem[];
    float*          chebs = (float*)(smem);                                  // [3][144][12]
    float*          Xs    = (float*)(smem + CHEB_BYTES);                     // [16][64][12]
    unsigned short* Ahi   = (unsigned short*)(smem + CHEB_BYTES + XS_BYTES);              // [144][72]
    unsigned short* Alo   = (unsigned short*)(smem + CHEB_BYTES + XS_BYTES + AHL_BYTES);  // [144][72]
    unsigned short* Whi   = (unsigned short*)(smem + CHEB_BYTES + XS_BYTES + 2*AHL_BYTES);            // [128][72]
    unsigned short* Wlo   = (unsigned short*)(smem + CHEB_BYTES + XS_BYTES + 2*AHL_BYTES + WHL_BYTES);// [128][72]

    const int t    = threadIdx.x;
    const int wv   = t >> 5;        // wave 0..8 -> C row tile
    const int lane = t & 31;
    const int l16  = lane & 15;
    const int hi   = lane >> 4;
    const int n0   = blockIdx.x * NT;
    const int b0   = blockIdx.y * BG;

    // Stage cheb rows: chebs[kp][row][j]; matrix index after view-scramble: m = kp*BATCH + b'.
    for (int idx = t; idx < 3 * MT * 9; idx += 288) {
        const int kp  = idx / (MT * 9);
        const int rem = idx % (MT * 9);
        const int row = rem / 9;
        const int j   = rem % 9;
        const int bl  = row / 9;
        const int i   = row % 9;
        chebs[(kp * MT + row) * CH_STRIDE + j] =
            chebraw[(size_t)(kp * BATCH + b0 + bl) * 81 + i * 9 + j];
    }

    v8f acc[8];
#pragma unroll
    for (int ct = 0; ct < 8; ++ct) acc[ct] = v8f{0, 0, 0, 0, 0, 0, 0, 0};

    for (int kc = 0; kc < INC / KT; ++kc) {   // 16 column chunks of X
        __syncthreads();
        // Stage X chunk transposed: Xs[b_l][c][j] (coalesced global reads)
#pragma unroll
        for (int s = 0; s < 32; ++s) {
            const int idx = t + 288 * s;      // 0..9215
            const int bl  = idx / (9 * KT);
            const int rem = idx % (9 * KT);
            const int j   = rem / KT;
            const int c   = rem % KT;
            Xs[(bl * KT + c) * XS_STRIDE + j] =
                X[((size_t)(b0 + bl) * NPT + j) * INC + kc * KT + c];
        }
        for (int kp = 0; kp < 3; ++kp) {      // cheb order = weight slab
            __syncthreads();
            // Stage W tile split hi/lo, transposed Whi/Wlo[n][c] (coalesced along n)
            for (int idx = t; idx < NT * (KT / 4); idx += 288) {
                const int nl = idx & 127;
                const int c4 = idx >> 7;                 // 0..15, 4 consecutive c
                const float* wp = &W[(size_t)kp * INC * OUTC +
                                     (size_t)(kc * KT + c4 * 4) * OUTC + n0 + nl];
                unsigned short h[4], l[4];
#pragma unroll
                for (int q = 0; q < 4; ++q) {
                    const float w = wp[(size_t)q * OUTC];
                    h[q] = bf16h(w);
                    l[q] = bf16h(w - bf16f(h[q]));
                }
                const int base = nl * AB_STRIDE + c4 * 4;
                *(unsigned*)&Whi[base]     = (unsigned)h[0] | ((unsigned)h[1] << 16);
                *(unsigned*)&Whi[base + 2] = (unsigned)h[2] | ((unsigned)h[3] << 16);
                *(unsigned*)&Wlo[base]     = (unsigned)l[0] | ((unsigned)l[1] << 16);
                *(unsigned*)&Wlo[base + 2] = (unsigned)l[2] | ((unsigned)l[3] << 16);
            }
            // Build A tile rows (fp32 dot-9), split hi/lo into LDS.
            {
                const int row = t >> 1;            // 0..143
                const int ch  = (t & 1) * 32;
                const int bl  = row / 9;
                const float* cb = &chebs[(kp * MT + row) * CH_STRIDE];
                const v4f  cb0 = *(const v4f*)(cb);
                const v4f  cb1 = *(const v4f*)(cb + 4);
                const float cb8 = cb[8];
                const float* xb = &Xs[bl * KT * XS_STRIDE];
#pragma unroll
                for (int c = 0; c < 32; c += 2) {
                    float a2[2];
#pragma unroll
                    for (int u = 0; u < 2; ++u) {
                        const float* xp = xb + (ch + c + u) * XS_STRIDE;
                        const v4f x0 = *(const v4f*)(xp);
                        const v4f x1 = *(const v4f*)(xp + 4);
                        a2[u] = cb0.x * x0.x + cb0.y * x0.y + cb0.z * x0.z + cb0.w * x0.w
                              + cb1.x * x1.x + cb1.y * x1.y + cb1.z * x1.z + cb1.w * x1.w
                              + cb8 * xp[8];
                    }
                    const unsigned short h0 = bf16h(a2[0]), h1 = bf16h(a2[1]);
                    const unsigned short l0 = bf16h(a2[0] - bf16f(h0));
                    const unsigned short l1 = bf16h(a2[1] - bf16f(h1));
                    const int base = row * AB_STRIDE + ch + c;
                    *(unsigned*)&Ahi[base] = (unsigned)h0 | ((unsigned)h1 << 16);
                    *(unsigned*)&Alo[base] = (unsigned)l0 | ((unsigned)l1 << 16);
                }
            }
            __syncthreads();
            // BF16x3 WMMA: two K=32 chunks per K_tile.
            const int rowA = wv * 16 + l16;
#pragma unroll
            for (int chunk = 0; chunk < 2; ++chunk) {
                // A frag (16-bit A 16x32 layout): lane half -> K = 8*hi..+7 and 16+8*hi..+7
                const int ka = chunk * 32 + 8 * hi;
                const v8bf ah0 = *(const v8bf*)&Ahi[rowA * AB_STRIDE + ka];
                const v8bf ah1 = *(const v8bf*)&Ahi[rowA * AB_STRIDE + ka + 16];
                const v8bf al0 = *(const v8bf*)&Alo[rowA * AB_STRIDE + ka];
                const v8bf al1 = *(const v8bf*)&Alo[rowA * AB_STRIDE + ka + 16];
                const v16bf a_hi = __builtin_shufflevector(ah0, ah1,
                    0,1,2,3,4,5,6,7,8,9,10,11,12,13,14,15);
                const v16bf a_lo = __builtin_shufflevector(al0, al1,
                    0,1,2,3,4,5,6,7,8,9,10,11,12,13,14,15);
                // B frag (16-bit B 32x16 layout): lane half -> K = 8*hi..+7 and 16+8*hi..+7
                const int kb = chunk * 32 + 8 * hi;
#pragma unroll
                for (int ct = 0; ct < 8; ++ct) {
                    const int nrow = ct * 16 + l16;
                    const v8bf bh0 = *(const v8bf*)&Whi[nrow * AB_STRIDE + kb];
                    const v8bf bh1 = *(const v8bf*)&Whi[nrow * AB_STRIDE + kb + 16];
                    const v8bf bl0 = *(const v8bf*)&Wlo[nrow * AB_STRIDE + kb];
                    const v8bf bl1 = *(const v8bf*)&Wlo[nrow * AB_STRIDE + kb + 16];
                    const v16bf b_hi = __builtin_shufflevector(bh0, bh1,
                        0,1,2,3,4,5,6,7,8,9,10,11,12,13,14,15);
                    const v16bf b_lo = __builtin_shufflevector(bl0, bl1,
                        0,1,2,3,4,5,6,7,8,9,10,11,12,13,14,15);
                    acc[ct] = __builtin_amdgcn_wmma_f32_16x16x32_bf16(
                        false, a_hi, false, b_hi, (short)0, acc[ct], false, false);
                    acc[ct] = __builtin_amdgcn_wmma_f32_16x16x32_bf16(
                        false, a_hi, false, b_lo, (short)0, acc[ct], false, false);
                    acc[ct] = __builtin_amdgcn_wmma_f32_16x16x32_bf16(
                        false, a_lo, false, b_hi, (short)0, acc[ct], false, false);
                }
            }
        }
    }

    // Epilogue: C layout (ISA 7.12.2): VGPR v -> row = 16*wv + v + 8*hi, col = ct*16 + l16
#pragma unroll
    for (int ct = 0; ct < 8; ++ct) {
        const int col = n0 + ct * 16 + l16;
        const float bv = bias[col];
#pragma unroll
        for (int v = 0; v < 8; ++v) {
            const int rl = wv * 16 + v + 8 * hi;
            out[(size_t)(blockIdx.y * MT + rl) * OUTC + col] = acc[ct][v] + bv;
        }
    }
}

// ---------------------------------------------------------------------------
extern "C" void kernel_launch(void* const* d_in, const int* in_sizes, int n_in,
                              void* d_out, int out_size, void* d_ws, size_t ws_size,
                              hipStream_t stream) {
    (void)in_sizes; (void)n_in; (void)out_size; (void)ws_size;
    const float* X    = (const float*)d_in[0];  // inputs   [8192,9,1024]
    const float* mul  = (const float*)d_in[2];  // mul_data [8192,9,9]
    const float* W    = (const float*)d_in[3];  // weight   [3,1,1024,1024]
    const float* bias = (const float*)d_in[4];  // bias     [1,1,1024]
    float* out = (float*)d_out;
    float* chebraw = (float*)d_ws;              // 3*8192*81 floats ~ 8 MB

    cheb_kernel<<<BATCH, 256, 0, stream>>>(mul, chebraw);

    const size_t shm = SMEM_BYTES;              // 148224 B (< 320 KB WGP LDS)
    hipFuncSetAttribute((const void*)cheb_gemm_kernel,
                        hipFuncAttributeMaxDynamicSharedMemorySize, (int)shm);
    dim3 grid(OUTC / NT, BATCH / BG);           // (8, 512)
    cheb_gemm_kernel<<<grid, 288, shm, stream>>>(X, W, bias, chebraw, out);
}
